// Histogram2D_28252294873506
// MI455X (gfx1250) — matrix operand order repaired and checked by reading, not compiled, not run on hardware
//
#include <hip/hip_runtime.h>

#define NBINS 128
#define KB 32
#define LSTRIDE 40  // halves per LDS row: 80B, 16B-aligned, bank-staggered

typedef __attribute__((ext_vector_type(16))) _Float16 v16h;
typedef __attribute__((ext_vector_type(8)))  _Float16 v8h;
typedef __attribute__((ext_vector_type(8)))  float    v8f;

__global__ void kde_zero_kernel(float* __restrict__ p, int n) {
    int i = blockIdx.x * blockDim.x + threadIdx.x;
    if (i < n) p[i] = 0.0f;
}

__global__ __launch_bounds__(256)
void kde_hist_wmma_kernel(const float* __restrict__ x,
                          const float* __restrict__ ex,
                          const float* __restrict__ ey,
                          float* __restrict__ hist,
                          int n, int nbatch) {
    // double-buffered f16 staging, transposed [bin][k]
    __shared__ __align__(16) _Float16 lkx[2][NBINS][LSTRIDE];
    __shared__ __align__(16) _Float16 lky[2][NBINS][LSTRIDE];

    const int tid  = threadIdx.x;
    const int lane = tid & 31;
    const int wave = tid >> 5;
    const int p_in = tid & 31;   // point within batch
    const int bgrp = tid >> 5;   // bin group: bins bgrp*16 .. +15

    // ---- hoist this thread's 16 bin centers (per axis) into registers
    float cx_r[16], cy_r[16];
#pragma unroll
    for (int i = 0; i < 16; ++i) {
        const int b = bgrp * 16 + i;
        cx_r[i] = 0.5f * (ex[b] + ex[b + 1]);
        cy_r[i] = 0.5f * (ey[b] + ey[b + 1]);
    }
    const float ibwx = 1.0f / (ex[1] - ex[0]);
    const float ibwy = 1.0f / (ey[1] - ey[0]);
    // exp(-0.5*(d/bw)^2) == exp2(c*d^2), c = -0.5*log2(e)/bw^2
    const float cfx = -0.72134752044448170f * ibwx * ibwx;
    const float cfy = -0.72134752044448170f * ibwy * ibwy;

    // per-wave accumulators: rows [wave*16, wave*16+16), 8 column tiles
    v8f acc[8];
#pragma unroll
    for (int t = 0; t < 8; ++t) acc[t] = (v8f){0,0,0,0,0,0,0,0};

    // branchless clamped point load; invalid lanes poisoned so exp2 -> 0
    auto load_pt = [&](int bi) -> float2 {
        int pg  = bi * KB + p_in;
        int pgc = (pg < n) ? pg : (n - 1);
        float2 v = *(const float2*)(x + (size_t)pgc * 6);
        if (pg >= n) { v.x = 1.0e30f; v.y = 1.0e30f; }
        return v;
    };

    const int mrow = wave * 16 + (lane & 15);
    const int akb  = (lane < 16) ? 0 : 8;
    const int bkb  = (lane < 16) ? 0 : 8;
    const int ncol0 = lane & 15;

    int bi  = blockIdx.x;
    int buf = 0;
    float2 xv = load_pt(bi < nbatch ? bi : 0);

    while (bi < nbatch) {
        const int bnext = bi + gridDim.x;
        float2 xv_next = load_pt(bnext < nbatch ? bnext : bi);  // prefetch

        // ---- phase 1: Gaussian kernels -> f16 LDS (buffer `buf`)
#pragma unroll
        for (int i = 0; i < 16; ++i) {
            const int bin = bgrp * 16 + i;
            const float dx = xv.x - cx_r[i];
            const float dy = xv.y - cy_r[i];
            lkx[buf][bin][p_in] = (_Float16)__builtin_amdgcn_exp2f(cfx * dx * dx);
            lky[buf][bin][p_in] = (_Float16)__builtin_amdgcn_exp2f(cfy * dy * dy);
        }
        __syncthreads();

        // ---- phase 2: rank-32 WMMA update
        // A fragment (16x32 f16): halves 0..7 = K=akb..+7, 8..15 = K=akb+16..+23
        v8h alo = *(const v8h*)&lkx[buf][mrow][akb];
        v8h ahi = *(const v8h*)&lkx[buf][mrow][akb + 16];
        v16h a = __builtin_shufflevector(alo, ahi,
                 0,1,2,3,4,5,6,7,8,9,10,11,12,13,14,15);

#pragma unroll
        for (int nt = 0; nt < 8; ++nt) {
            const int ncol = nt * 16 + ncol0;
            v8h blo = *(const v8h*)&lky[buf][ncol][bkb];
            v8h bhi = *(const v8h*)&lky[buf][ncol][bkb + 16];
            v16h b = __builtin_shufflevector(blo, bhi,
                     0,1,2,3,4,5,6,7,8,9,10,11,12,13,14,15);
            acc[nt] = __builtin_amdgcn_wmma_f32_16x16x32_f16(
                false, a, false, b, (short)0, acc[nt], false, false);
        }
        // no trailing barrier: next iteration writes the other buffer,
        // and per-wave s_wait_dscnt before the WMMAs orders this buffer's reads
        buf ^= 1;
        xv = xv_next;
        bi = bnext;
    }

    // ---- flush: C/D layout: VGPR r -> M = r + (lane<16?0:8), N = lane&15
    const int msub = (lane < 16) ? 0 : 8;
#pragma unroll
    for (int nt = 0; nt < 8; ++nt) {
#pragma unroll
        for (int r = 0; r < 8; ++r) {
            atomicAdd(&hist[(wave * 16 + msub + r) * NBINS + nt * 16 + ncol0],
                      acc[nt][r]);
        }
    }
}

__global__ __launch_bounds__(256)
void kde_finalize_kernel(const float* __restrict__ hist,
                         float* __restrict__ out,
                         const float* __restrict__ ex,
                         const float* __restrict__ ey) {
    __shared__ float red[256];
    const int tid = threadIdx.x;
    float s = 0.0f;
    for (int i = tid; i < NBINS * NBINS; i += 256) s += hist[i];
    red[tid] = s;
    __syncthreads();
    for (int off = 128; off > 0; off >>= 1) {
        if (tid < off) red[tid] += red[tid + off];
        __syncthreads();
    }
    const float bwx = ex[1] - ex[0];
    const float bwy = ey[1] - ey[0];
    const float scale = 1.0f / (red[0] * bwx * bwy);
    for (int i = tid; i < NBINS * NBINS; i += 256) out[i] = hist[i] * scale;
}

extern "C" void kernel_launch(void* const* d_in, const int* in_sizes, int n_in,
                              void* d_out, int out_size, void* d_ws, size_t ws_size,
                              hipStream_t stream) {
    const float* x  = (const float*)d_in[0];   // (500000, 6) f32
    const float* ex = (const float*)d_in[1];   // (129,) f32
    const float* ey = (const float*)d_in[2];   // (129,) f32
    float* out  = (float*)d_out;               // (128,128) f32
    float* hist = (float*)d_ws;                // f32 accumulator

    const int n      = in_sizes[0] / 6;
    const int nbatch = (n + KB - 1) / KB;

    kde_zero_kernel<<<(NBINS * NBINS + 255) / 256, 256, 0, stream>>>(hist, NBINS * NBINS);

    int blocks = 512;
    if (blocks > nbatch) blocks = nbatch;
    kde_hist_wmma_kernel<<<blocks, 256, 0, stream>>>(x, ex, ey, hist, n, nbatch);

    kde_finalize_kernel<<<1, 256, 0, stream>>>(hist, out, ex, ey);
}
